// AutoEncoder_16630113370691
// MI455X (gfx1250) — matrix operand order repaired and checked by reading, not compiled, not run on hardware
//
#include <hip/hip_runtime.h>

typedef __attribute__((ext_vector_type(16))) __bf16        v16bf;
typedef __attribute__((ext_vector_type(8)))  float         v8f;
typedef __attribute__((ext_vector_type(8)))  unsigned int  v8u;

union Frag { v8u u; v16bf v; };

#define ROWS 32   // rows per workgroup for the batched GEMM kernels

__device__ __forceinline__ unsigned short f2bf(float f) {
  union { float f; unsigned u; } x; x.f = f;
  unsigned r = x.u + 0x7fffu + ((x.u >> 16) & 1u);   // round-to-nearest-even
  return (unsigned short)(r >> 16);
}
__device__ __forceinline__ float bf2f(unsigned short h) {
  union { unsigned u; float f; } x; x.u = ((unsigned)h) << 16;
  return x.f;
}
__device__ __forceinline__ float lrelu(float x) { return x >= 0.f ? x : 0.2f * x; }
__device__ __forceinline__ float sigm(float x)  { return 1.f / (1.f + __expf(-x)); }

// ---- A fragment (16x32 bf16, M x K) from LDS, row-major activations ----
// lane L: row M = L%16, half = L/16 ; element e holds K = (e>>3)*16 + half*8 + (e&7)
__device__ __forceinline__ v16bf load_a(const unsigned short* X, int stride,
                                        int rowbase, int kbase) {
  const int lane = threadIdx.x & 31;
  const int m = lane & 15, half = lane >> 4;
  const unsigned short* p = X + (rowbase + m) * stride + kbase + half * 8;
  uint4 lo = *(const uint4*)p;          // K = kbase+half*8   .. +7
  uint4 hi = *(const uint4*)(p + 16);   // K = kbase+16+half*8.. +7
  Frag f; f.u = (v8u){lo.x, lo.y, lo.z, lo.w, hi.x, hi.y, hi.z, hi.w};
  return f.v;
}

// ---- B fragment (32x16 bf16, K x N) from weights stored [out][in] ----
// lane L: col N = L%16, half = L/16 ; element e holds K = kbase + (e>>3)*16 + half*8 + (e&7)
__device__ __forceinline__ v16bf load_b(const unsigned short* W, int stride,
                                        int nbase, int kbase) {
  const int lane = threadIdx.x & 31;
  const int n = lane & 15, half = lane >> 4;
  const unsigned short* p = W + (size_t)(nbase + n) * stride + kbase + half * 8;
  uint4 lo = *(const uint4*)p;
  uint4 hi = *(const uint4*)(p + 16);
  Frag f; f.u = (v8u){lo.x, lo.y, lo.z, lo.w, hi.x, hi.y, hi.z, hi.w};
  return f.v;
}

// ---- A fragment for GEMV: vector in row 0, rows 1..15 zero ----
__device__ __forceinline__ v16bf load_a_vec(const unsigned short* x, int kbase) {
  const int lane = threadIdx.x & 31;
  const int half = lane >> 4;
  Frag f;
  if ((lane & 15) == 0) {
    const unsigned short* p = x + kbase + half * 8;
    uint4 lo = *(const uint4*)p;
    uint4 hi = *(const uint4*)(p + 16);
    f.u = (v8u){lo.x, lo.y, lo.z, lo.w, hi.x, hi.y, hi.z, hi.w};
  } else {
    f.u = (v8u){0, 0, 0, 0, 0, 0, 0, 0};
  }
  return f.v;
}

// cooperative prefetch of a weight matrix into caches (global_prefetch_b8)
__device__ __forceinline__ void prefetch_w(const unsigned short* W, int nush) {
  for (int i = threadIdx.x * 64; i < nush; i += 256 * 64)
    __builtin_prefetch(W + i, 0, 3);
}

// One GEMM stage: OUT = epilogue( X[ROWS,XK] @ W^T + bias ), 128 output cols.
// 8 waves; wave == output column tile; B loaded once per K-chunk, reused for
// both row tiles (two accumulators).
__device__ void gemm_stage(const unsigned short* X, int XK,
                           const unsigned short* W, const float* bias,
                           bool act_lrelu,
                           const unsigned short* RES,     // nullable, LDS stride 128
                           unsigned short* OUT_LDS,       // nullable, LDS stride 128
                           unsigned short* OUT_G_BF,      // nullable, global bf16 (rowbase applied)
                           float* OUT_G_F32,              // nullable, global f32  (rowbase applied)
                           float* OUT_LDS_F32)            // nullable, LDS f32 stride 128
{
  const int nt   = threadIdx.x >> 5;          // wave id == column tile
  const int lane = threadIdx.x & 31;
  const int n = lane & 15, half = lane >> 4;
  const int kch = XK >> 5;
  v8f c0 = {0.f, 0.f, 0.f, 0.f, 0.f, 0.f, 0.f, 0.f};
  v8f c1 = {0.f, 0.f, 0.f, 0.f, 0.f, 0.f, 0.f, 0.f};
  for (int kc = 0; kc < kch; ++kc) {
    v16bf b  = load_b(W, XK, nt * 16, kc * 32);
    v16bf a0 = load_a(X, XK, 0,  kc * 32);
    v16bf a1 = load_a(X, XK, 16, kc * 32);
    c0 = __builtin_amdgcn_wmma_f32_16x16x32_bf16(false, a0, false, b, (short)0, c0, false, false);
    c1 = __builtin_amdgcn_wmma_f32_16x16x32_bf16(false, a1, false, b, (short)0, c1, false, false);
  }
  const int col = nt * 16 + n;
  const float bv = bias ? bias[col] : 0.f;
#pragma unroll
  for (int rt = 0; rt < 2; ++rt) {
    v8f c = rt ? c1 : c0;
#pragma unroll
    for (int r = 0; r < 8; ++r) {
      const int row = rt * 16 + r + half * 8;
      float v = c[r] + bv;
      if (RES) v += bf2f(RES[row * 128 + col]);
      if (act_lrelu) v = lrelu(v);
      if (OUT_LDS)      OUT_LDS[row * 128 + col] = f2bf(v);
      if (OUT_G_BF)     OUT_G_BF[(size_t)row * 128 + col] = f2bf(v);
      if (OUT_G_F32)    OUT_G_F32[(size_t)row * 128 + col] = v;
      if (OUT_LDS_F32)  OUT_LDS_F32[row * 128 + col] = v;
    }
  }
}

// ================= prep kernels =================
__global__ void k_f2bf(const float* __restrict__ src, unsigned short* __restrict__ dst, int n) {
  int i = blockIdx.x * 256 + threadIdx.x;
  if (i < n) dst[i] = f2bf(src[i]);
}
__global__ void k_zero(float* p, int n) {
  int i = blockIdx.x * 256 + threadIdx.x;
  if (i < n) p[i] = 0.f;
}
// deterministic column-sum of per-block partials: parts[nb][128] -> out[128]
__global__ void k_colsum(const float* __restrict__ parts, float* __restrict__ out, int nb) {
  int t = threadIdx.x;
  if (t < 128) {
    float s = 0.f;
    for (int b = 0; b < nb; ++b) s += parts[(size_t)b * 128 + t];
    out[t] = s;
  }
}

// ================= encoder stage 1 =================
__global__ __launch_bounds__(256) void k_enc1(
    const int* __restrict__ atom, const float* __restrict__ time_,
    const float* __restrict__ mag, const float* __restrict__ atom_emb,
    const float* __restrict__ mag_w, const float* __restrict__ mag_b,
    const float* __restrict__ pos_w, const float* __restrict__ pos_b,
    const unsigned short* __restrict__ w_reduce, const float* __restrict__ reduce_b,
    const unsigned short* __restrict__ w_net1, const float* __restrict__ net_b1,
    const unsigned short* __restrict__ w_net2, const float* __restrict__ net_b2,
    const unsigned short* __restrict__ w_final, const float* __restrict__ final_b,
    unsigned short* __restrict__ x_enc, float* __restrict__ gparts)
{
  __shared__ unsigned short A0[ROWS * 384];
  __shared__ unsigned short A1[ROWS * 128];
  __shared__ unsigned short A2[ROWS * 128];
  __shared__ float FOUT[ROWS * 128];
  const int tid = threadIdx.x;
  const size_t rowbase = (size_t)blockIdx.x * ROWS;

  // build concat([ae, te, me]) in bf16
  for (int idx = tid; idx < ROWS * 384; idx += 256) {
    const int r = idx / 384, c = idx % 384;
    const size_t grow = rowbase + r;
    float v;
    if (c < 128)      v = atom_emb[(size_t)atom[grow] * 128 + c];
    else if (c < 256) v = time_[grow] * pos_w[c - 128] + pos_b[c - 128];
    else              v = mag[grow] * mag_w[c - 256] + mag_b[c - 256];
    A0[idx] = f2bf(v);
  }
  prefetch_w(w_reduce, 49152);
  __syncthreads();
  gemm_stage(A0, 384, w_reduce, reduce_b, false, nullptr, A1, nullptr, nullptr, nullptr);
  prefetch_w(w_net1, 16384);
  __syncthreads();
  for (int l = 0; l < 4; ++l) {
    gemm_stage(A1, 128, w_net1 + l * 16384, net_b1 + l * 128, true, nullptr, A2, nullptr, nullptr, nullptr);
    prefetch_w(w_net2 + l * 16384, 16384);
    __syncthreads();
    gemm_stage(A2, 128, w_net2 + l * 16384, net_b2 + l * 128, true, A1, A1, nullptr, nullptr, nullptr);
    __syncthreads();
  }
  gemm_stage(A1, 128, w_final, final_b, false, nullptr, nullptr,
             x_enc + rowbase * 128, nullptr, FOUT);
  __syncthreads();
  if (tid < 128) {                 // deterministic per-block column sum
    float s = 0.f;
    for (int r = 0; r < ROWS; ++r) s += FOUT[r * 128 + tid];
    gparts[(size_t)blockIdx.x * 128 + tid] = s;
  }
}

// ================= encoder stage 2 =================
__global__ __launch_bounds__(256) void k_enc2(
    const unsigned short* __restrict__ x_enc, const float* __restrict__ g_vec,
    const unsigned short* __restrict__ w_redg, const float* __restrict__ redg_b,
    const unsigned short* __restrict__ w_wg1, const float* __restrict__ wg_b1,
    const unsigned short* __restrict__ w_wg2, const float* __restrict__ wg_b2,
    const unsigned short* __restrict__ w_gfinal, const float* __restrict__ gfinal_b,
    float* __restrict__ zparts)
{
  __shared__ unsigned short A0[ROWS * 256];
  __shared__ unsigned short A1[ROWS * 128];
  __shared__ unsigned short A2[ROWS * 128];
  __shared__ float FOUT[ROWS * 128];
  const int tid = threadIdx.x;
  const size_t rowbase = (size_t)blockIdx.x * ROWS;

  for (int idx = tid; idx < ROWS * 256; idx += 256) {
    const int r = idx >> 8, c = idx & 255;
    A0[idx] = (c < 128) ? x_enc[(rowbase + r) * 128 + c] : f2bf(g_vec[c - 128]);
  }
  prefetch_w(w_redg, 32768);
  __syncthreads();
  gemm_stage(A0, 256, w_redg, redg_b, false, nullptr, A1, nullptr, nullptr, nullptr);
  __syncthreads();
  for (int l = 0; l < 4; ++l) {
    gemm_stage(A1, 128, w_wg1 + l * 16384, wg_b1 + l * 128, true, nullptr, A2, nullptr, nullptr, nullptr);
    __syncthreads();
    gemm_stage(A2, 128, w_wg2 + l * 16384, wg_b2 + l * 128, true, A1, A1, nullptr, nullptr, nullptr);
    __syncthreads();
  }
  gemm_stage(A1, 128, w_gfinal, gfinal_b, false, nullptr, nullptr, nullptr, nullptr, FOUT);
  __syncthreads();
  if (tid < 128) {
    float s = 0.f;
    for (int r = 0; r < ROWS; ++r) s += FOUT[r * 128 + tid];
    zparts[(size_t)blockIdx.x * 128 + tid] = s;
  }
}

// ================= GRU decoder =================
// One persistent workgroup, 16 waves. Weight residency (MI455X budget):
//   layer 0 : WMMA B-fragments register-resident on waves 0-7 (192 VGPR/lane)
//   layer 1 : register-resident on waves 8-15
//   layer 2 : LDS-resident (ih+hh, 196.6 KB)
//   layer 3 : ih LDS-resident (98.3 KB), hh streamed from L2 (only global traffic)
// Per layer: 48 GEMV tile-tasks (24 ih + 24 hh), 6 per wave of the active group,
// 4 K-chunks each; vector lives in row 0 of the A fragment.
#define GRU_MATVEC(GETB)                                                        \
  _Pragma("unroll")                                                             \
  for (int j = 0; j < 6; ++j) {                                                 \
    const int tt = slot * 6 + j;                                                \
    const bool is_hh = tt >= 24;                                                \
    const int nt = is_hh ? tt - 24 : tt;                                        \
    const unsigned short* xv = is_hh ? hb[l] : xb;                              \
    v8f c = {0.f, 0.f, 0.f, 0.f, 0.f, 0.f, 0.f, 0.f};                           \
    _Pragma("unroll")                                                           \
    for (int kc = 0; kc < 4; ++kc) {                                            \
      v16bf a = load_a_vec(xv, kc * 32);                                        \
      v16bf b = (GETB);                                                         \
      c = __builtin_amdgcn_wmma_f32_16x16x32_bf16(false, a, false, b,           \
                                                  (short)0, c, false, false);   \
    }                                                                           \
    float* dst = is_hh ? gh : gi;                                               \
    if (lane < 16) dst[nt * 16 + lane] = c[0];                                  \
  }

__global__ __launch_bounds__(512) void k_gru(
    const float* __restrict__ z_vec,
    const unsigned short* __restrict__ w_ih, const unsigned short* __restrict__ w_hh,
    const float* __restrict__ bih, const float* __restrict__ bhh,
    unsigned short* __restrict__ enc_out, float* __restrict__ z_out, int n_steps)
{
  __shared__ unsigned short w2lds[2 * 49152];  // layer2 ih (0..) + hh (+49152)
  __shared__ unsigned short w3ih[49152];       // layer3 ih
  __shared__ unsigned short xb[128];
  __shared__ unsigned short hb[4][128];
  __shared__ float hf[4][128];
  __shared__ float gi[384];
  __shared__ float gh[384];
  __shared__ float sbih[1536];
  __shared__ float sbhh[1536];
  const int tid = threadIdx.x;
  const int wave = tid >> 5, lane = tid & 31;
  const unsigned short* w_hh3 = w_hh + (size_t)3 * 49152;

  // stage layer2 (ih+hh) and layer3-ih weights into LDS
  for (int i = tid; i < 49152 / 8; i += 512) {
    ((uint4*)w2lds)[i]           = ((const uint4*)(w_ih + 2 * 49152))[i];
    ((uint4*)(w2lds + 49152))[i] = ((const uint4*)(w_hh + 2 * 49152))[i];
    ((uint4*)w3ih)[i]            = ((const uint4*)(w_ih + 3 * 49152))[i];
  }
  for (int i = tid; i < 1536; i += 512) { sbih[i] = bih[i]; sbhh[i] = bhh[i]; }

  // register-resident fragments: waves 0-7 hold layer 0, waves 8-15 layer 1
  v16bf wf[24];
  {
    const int lay  = wave >> 3;        // 0 or 1
    const int slot = wave & 7;
#pragma unroll
    for (int j = 0; j < 6; ++j) {
      const int tt = slot * 6 + j;
      const bool is_hh = tt >= 24;
      const int nt = is_hh ? tt - 24 : tt;
      const unsigned short* W = (is_hh ? w_hh : w_ih) + (size_t)lay * 49152;
#pragma unroll
      for (int kc = 0; kc < 4; ++kc)
        wf[j * 4 + kc] = load_b(W, 128, nt * 16, kc * 32);
    }
  }

  if (tid < 128) {
    const float zv = z_vec[tid];
    xb[tid] = f2bf(zv);
    z_out[tid] = zv;                   // z is also a model output
    for (int l = 0; l < 4; ++l) { hf[l][tid] = 0.f; hb[l][tid] = 0; }
  }
  __syncthreads();

  for (int step = 0; step < n_steps; ++step) {
#pragma unroll
    for (int l = 0; l < 4; ++l) {
      if (l == 0 && wave < 8) {
        const int slot = wave;
        GRU_MATVEC(wf[j * 4 + kc])
      } else if (l == 1 && wave >= 8) {
        const int slot = wave - 8;
        GRU_MATVEC(wf[j * 4 + kc])
      } else if (l == 2 && wave < 8) {
        const int slot = wave;
        GRU_MATVEC(load_b(is_hh ? w2lds + 49152 : w2lds, 128, nt * 16, kc * 32))
      } else if (l == 3 && wave >= 8) {
        const int slot = wave - 8;
        GRU_MATVEC(is_hh ? load_b(w_hh3, 128, nt * 16, kc * 32)
                         : load_b(w3ih, 128, nt * 16, kc * 32))
      }
      __syncthreads();
      if (tid < 128) {
        const int t = tid;
        const float r  = sigm(gi[t]       + sbih[l*384 + t]       + gh[t]       + sbhh[l*384 + t]);
        const float zt = sigm(gi[128 + t] + sbih[l*384 + 128 + t] + gh[128 + t] + sbhh[l*384 + 128 + t]);
        const float nn = tanhf(gi[256 + t] + sbih[l*384 + 256 + t]
                               + r * (gh[256 + t] + sbhh[l*384 + 256 + t]));
        const float h = (1.f - zt) * nn + zt * hf[l][t];
        hf[l][t] = h;
        const unsigned short h16 = f2bf(h);
        hb[l][t] = h16;
        xb[t] = h16;                                 // next layer input / feedback
        if (l == 3) enc_out[(size_t)step * 128 + t] = h16;
      }
      __syncthreads();
    }
  }
}

// ================= output heads =================
__global__ __launch_bounds__(256) void k_heads(
    const unsigned short* __restrict__ enc,
    const unsigned short* __restrict__ w_ah1, const float* __restrict__ ah_b1,
    const unsigned short* __restrict__ w_ah2, const float* __restrict__ ah_b2,
    const unsigned short* __restrict__ w_ah,  const float* __restrict__ ah_b,
    const unsigned short* __restrict__ w_ph1, const float* __restrict__ ph_b1,
    const unsigned short* __restrict__ w_ph2, const float* __restrict__ ph_b2,
    const float* __restrict__ ph_w, const float* __restrict__ ph_b,
    const unsigned short* __restrict__ w_mh1, const float* __restrict__ mh_b1,
    const unsigned short* __restrict__ w_mh2, const float* __restrict__ mh_b2,
    const float* __restrict__ mh_w, const float* __restrict__ mh_b,
    float* __restrict__ atoms_out, float* __restrict__ pos_out, float* __restrict__ mags_out)
{
  __shared__ unsigned short E [ROWS * 128];
  __shared__ unsigned short A1[ROWS * 128];
  __shared__ unsigned short A2[ROWS * 128];
  const int tid = threadIdx.x;
  const size_t rowbase = (size_t)blockIdx.x * ROWS;

  for (int idx = tid; idx < ROWS * 128; idx += 256) E[idx] = enc[rowbase * 128 + idx];

  // ---- atoms head: 2 res blocks + [128,128] final ----
  __syncthreads();
  for (int idx = tid; idx < ROWS * 128; idx += 256) A1[idx] = E[idx];
  __syncthreads();
  for (int l = 0; l < 2; ++l) {
    gemm_stage(A1, 128, w_ah1 + l * 16384, ah_b1 + l * 128, true, nullptr, A2, nullptr, nullptr, nullptr);
    __syncthreads();
    gemm_stage(A2, 128, w_ah2 + l * 16384, ah_b2 + l * 128, true, A1, A1, nullptr, nullptr, nullptr);
    __syncthreads();
  }
  gemm_stage(A1, 128, w_ah, ah_b, false, nullptr, nullptr, nullptr,
             atoms_out + rowbase * 128, nullptr);
  __syncthreads();

  // ---- pos head: 2 res blocks + [1,128] final (scalar dot), clipped ----
  for (int idx = tid; idx < ROWS * 128; idx += 256) A1[idx] = E[idx];
  __syncthreads();
  for (int l = 0; l < 2; ++l) {
    gemm_stage(A1, 128, w_ph1 + l * 16384, ph_b1 + l * 128, true, nullptr, A2, nullptr, nullptr, nullptr);
    __syncthreads();
    gemm_stage(A2, 128, w_ph2 + l * 16384, ph_b2 + l * 128, true, A1, A1, nullptr, nullptr, nullptr);
    __syncthreads();
  }
  if (tid < ROWS) {
    float s = ph_b[0];
    for (int k = 0; k < 128; ++k) s += bf2f(A1[tid * 128 + k]) * ph_w[k];
    pos_out[rowbase + tid] = fminf(fmaxf(s, 0.f), 1.f);
  }
  __syncthreads();

  // ---- mag head ----
  for (int idx = tid; idx < ROWS * 128; idx += 256) A1[idx] = E[idx];
  __syncthreads();
  for (int l = 0; l < 2; ++l) {
    gemm_stage(A1, 128, w_mh1 + l * 16384, mh_b1 + l * 128, true, nullptr, A2, nullptr, nullptr, nullptr);
    __syncthreads();
    gemm_stage(A2, 128, w_mh2 + l * 16384, mh_b2 + l * 128, true, A1, A1, nullptr, nullptr, nullptr);
    __syncthreads();
  }
  if (tid < ROWS) {
    float s = mh_b[0];
    for (int k = 0; k < 128; ++k) s += bf2f(A1[tid * 128 + k]) * mh_w[k];
    mags_out[rowbase + tid] = s;
  }
}

// ================= host launch =================
extern "C" void kernel_launch(void* const* d_in, const int* in_sizes, int n_in,
                              void* d_out, int out_size, void* d_ws, size_t ws_size,
                              hipStream_t stream) {
  const int N = in_sizes[0];          // 32768
  const int NB = N / ROWS;            // blocks for row-tiled kernels

  char* ws = (char*)d_ws;
  float* g_acc = (float*)ws;                       // [128]
  float* z_acc = (float*)(ws + 512);               // [128]
  unsigned short* wbf = (unsigned short*)(ws + 1024);

  // bf16 weight region offsets (ushort units)
  unsigned short* w_reduce = wbf + 0;        // 128x384
  unsigned short* w_net1   = wbf + 49152;    // 4x128x128
  unsigned short* w_net2   = wbf + 114688;
  unsigned short* w_final  = wbf + 180224;   // 128x128
  unsigned short* w_redg   = wbf + 196608;   // 128x256
  unsigned short* w_wg1    = wbf + 229376;
  unsigned short* w_wg2    = wbf + 294912;
  unsigned short* w_gfinal = wbf + 360448;
  unsigned short* w_gih    = wbf + 376832;   // 4x384x128
  unsigned short* w_ghh    = wbf + 573440;
  unsigned short* w_ah1    = wbf + 770048;   // 2x128x128
  unsigned short* w_ah2    = wbf + 802816;
  unsigned short* w_ah     = wbf + 835584;
  unsigned short* w_ph1    = wbf + 851968;
  unsigned short* w_ph2    = wbf + 884736;
  unsigned short* w_mh1    = wbf + 917504;
  unsigned short* w_mh2    = wbf + 950272;
  const size_t W_END = 983040;               // ushorts

  unsigned short* x_enc = (unsigned short*)(ws + 1024 + W_END * 2);
  unsigned short* enc   = x_enc + (size_t)N * 128;
  float* gparts = (float*)((char*)(enc + (size_t)N * 128));
  float* zparts = gparts + (size_t)NB * 128;

  float* atoms_out = (float*)d_out;
  float* pos_out   = atoms_out + (size_t)N * 128;
  float* mags_out  = pos_out + N;
  float* z_out     = mags_out + N;

  // zero g/z accumulators (contiguous 256 floats)
  k_zero<<<1, 256, 0, stream>>>(g_acc, 256);

  auto cv = [&](int idx, unsigned short* dst, int n) {
    k_f2bf<<<(n + 255) / 256, 256, 0, stream>>>((const float*)d_in[idx], dst, n);
  };
  cv(8,  w_reduce, 49152);
  cv(10, w_net1,   65536);
  cv(12, w_net2,   65536);
  cv(14, w_final,  16384);
  cv(16, w_redg,   32768);
  cv(18, w_wg1,    65536);
  cv(20, w_wg2,    65536);
  cv(22, w_gfinal, 16384);
  cv(24, w_gih,    196608);
  cv(25, w_ghh,    196608);
  cv(28, w_ah1,    32768);
  cv(30, w_ah2,    32768);
  cv(32, w_ah,     16384);
  cv(34, w_ph1,    32768);
  cv(36, w_ph2,    32768);
  cv(40, w_mh1,    32768);
  cv(42, w_mh2,    32768);

  k_enc1<<<NB, 256, 0, stream>>>(
      (const int*)d_in[0], (const float*)d_in[1], (const float*)d_in[2],
      (const float*)d_in[3], (const float*)d_in[4], (const float*)d_in[5],
      (const float*)d_in[6], (const float*)d_in[7],
      w_reduce, (const float*)d_in[9],
      w_net1, (const float*)d_in[11],
      w_net2, (const float*)d_in[13],
      w_final, (const float*)d_in[15],
      x_enc, gparts);
  k_colsum<<<1, 128, 0, stream>>>(gparts, g_acc, NB);

  k_enc2<<<NB, 256, 0, stream>>>(
      x_enc, g_acc,
      w_redg, (const float*)d_in[17],
      w_wg1, (const float*)d_in[19],
      w_wg2, (const float*)d_in[21],
      w_gfinal, (const float*)d_in[23],
      zparts);
  k_colsum<<<1, 128, 0, stream>>>(zparts, z_acc, NB);

  k_gru<<<1, 512, 0, stream>>>(
      z_acc, w_gih, w_ghh,
      (const float*)d_in[26], (const float*)d_in[27],
      enc, z_out, N);

  k_heads<<<NB, 256, 0, stream>>>(
      enc,
      w_ah1, (const float*)d_in[29], w_ah2, (const float*)d_in[31],
      w_ah,  (const float*)d_in[33],
      w_ph1, (const float*)d_in[35], w_ph2, (const float*)d_in[37],
      (const float*)d_in[38], (const float*)d_in[39],
      w_mh1, (const float*)d_in[41], w_mh2, (const float*)d_in[43],
      (const float*)d_in[44], (const float*)d_in[45],
      atoms_out, pos_out, mags_out);
}
